// Decoder_20298015441034
// MI455X (gfx1250) — matrix operand order repaired and checked by reading, not compiled, not run on hardware
//
#include <hip/hip_runtime.h>
#include <hip/hip_bf16.h>

// ---------------------------------------------------------------------------
// Sizes (fixed by the reference)
// ---------------------------------------------------------------------------
#define Hh   1024
#define Ee   512
#define Ss   128
#define Ii   32
#define Bb   64
#define Tt   64
#define K_IH 2560            // E + 2H
#define K_C  3584            // E + 2H + H   (fused [W_ih | W_hh])
#define G4H  4096            // 4*H
#define NEGV (-1.0e12f)

typedef __bf16 bf16_t;
typedef __attribute__((ext_vector_type(16))) __bf16 v16bf;
typedef __attribute__((ext_vector_type(8)))  __bf16 v8bf;
typedef __attribute__((ext_vector_type(8)))  float  v8f;

__device__ __forceinline__ float sigf(float x) { return 1.0f / (1.0f + __expf(-x)); }

// ---------------------------------------------------------------------------
// Register-blocked bf16 WMMA GEMM:
//   C[M][N] = A[M][K] * B[N][K]^T + bias[N]   (f32 accumulate)
// Each wave computes an MT x NT grid of 16x16 tiles; A fragments are reused
// across NT column tiles and B fragments across MT row tiles, so the loop
// issues MT+NT fragment loads per MT*NT WMMAs (1 load/WMMA at 2x2 vs 2 at
// 1x1) -- this matters because the per-step GEMM is the sequential critical
// path and is fed from L2-resident weights.
// grid = (N/(16*NT*8), M/(16*MT)), block = 256 (8 waves).
// A-fragment (16x32 bf16): lane<16 -> row=lane, K {0..7,16..23};
//                          lane>=16 -> row=lane-16, K {8..15,24..31}.
// B-fragment (32x16 bf16): col=lane&15, K same as A-fragment {0..7,16..23}/{8..15,24..31}.
// Both are two global_load_b128 per fragment from row-major bf16 arrays.
// ---------------------------------------------------------------------------
template <int MT, int NT>
__global__ __launch_bounds__(256) void gemm_bf16_wmma(
    const bf16_t* __restrict__ A, int lda,
    const bf16_t* __restrict__ Bw, int ldb,
    const float*  __restrict__ bias,
    float* __restrict__ C, int N, int K)
{
    const int lane = threadIdx.x & 31;
    const int wave = threadIdx.x >> 5;
    const int lo = lane & 15;
    const int hi = lane >> 4;
    const int n0 = (blockIdx.x * 8 + wave) * (16 * NT);
    const int m0 = blockIdx.y * (16 * MT);

    const bf16_t* arow[MT];
    const bf16_t* brow[NT];
#pragma unroll
    for (int i = 0; i < MT; ++i)
        arow[i] = A + (size_t)(m0 + i * 16 + lo) * lda + hi * 8;
#pragma unroll
    for (int j = 0; j < NT; ++j)
        brow[j] = Bw + (size_t)(n0 + j * 16 + lo) * ldb + hi * 8;

    v8f acc[MT][NT] = {};

#pragma unroll 2
    for (int kb = 0; kb < K; kb += 32) {
        union Frag { v16bf v; v8bf h[2]; };
        Frag a[MT], b[NT];
#pragma unroll
        for (int i = 0; i < MT; ++i) {
            a[i].h[0] = *(const v8bf*)(arow[i] + kb);
            a[i].h[1] = *(const v8bf*)(arow[i] + kb + 16);
        }
#pragma unroll
        for (int j = 0; j < NT; ++j) {
            b[j].h[0] = *(const v8bf*)(brow[j] + kb);
            b[j].h[1] = *(const v8bf*)(brow[j] + kb + 16);
            // Pull the next K-chunk of this weight row toward the WGP while
            // the WMMAs below retire (global_prefetch_b8; speculative, OOB ok).
            __builtin_prefetch(brow[j] + kb + 128, 0, 1);
        }
#pragma unroll
        for (int i = 0; i < MT; ++i)
#pragma unroll
            for (int j = 0; j < NT; ++j)
                acc[i][j] = __builtin_amdgcn_wmma_f32_16x16x32_bf16(
                    false, a[i].v, false, b[j].v, (short)0, acc[i][j],
                    false, false);
    }

#pragma unroll
    for (int i = 0; i < MT; ++i) {
        const int r0 = m0 + i * 16 + hi * 8;        // element v -> row r0+v
#pragma unroll
        for (int j = 0; j < NT; ++j) {
            const int col = n0 + j * 16 + lo;
            const float bv = bias ? bias[col] : 0.0f;
#pragma unroll
            for (int v = 0; v < 8; ++v)
                C[(size_t)(r0 + v) * N + col] = acc[i][j][v] + bv;
        }
    }
}

// ---------------------------------------------------------------------------
// One-time weight preparation
// ---------------------------------------------------------------------------
__global__ void build_wc_kernel(const float* __restrict__ W_ih,
                                const float* __restrict__ W_hh,
                                bf16_t* __restrict__ Wc)
{
    const size_t total = (size_t)G4H * K_C;
    for (size_t i = blockIdx.x * 256ull + threadIdx.x; i < total;
         i += (size_t)gridDim.x * 256ull) {
        int n = (int)(i / K_C), k = (int)(i % K_C);
        float v = (k < K_IH) ? W_ih[(size_t)n * K_IH + k]
                             : W_hh[(size_t)n * Hh + (k - K_IH)];
        Wc[i] = (bf16_t)v;
    }
}

__global__ void bias_kernel(const float* a, const float* b, float* o, int n) {
    int i = blockIdx.x * 256 + threadIdx.x;
    if (i < n) o[i] = a[i] + b[i];
}

__global__ void cvt_kernel(const float* __restrict__ src,
                           bf16_t* __restrict__ dst, size_t n) {
    for (size_t i = blockIdx.x * 256ull + threadIdx.x; i < n;
         i += (size_t)gridDim.x * 256ull)
        dst[i] = (bf16_t)src[i];
}

// ---------------------------------------------------------------------------
// Recurrent-state init: xh = [emb0 | ctx0 | aligned | h], hc = [h | ctx]
// ---------------------------------------------------------------------------
__global__ void init_kernel(const int* __restrict__ input_ids,
                            const float* __restrict__ context,
                            const float* __restrict__ embedding,
                            bf16_t* __restrict__ xh, bf16_t* __restrict__ hc,
                            float* __restrict__ h, float* __restrict__ c)
{
    const int b = blockIdx.x, tid = threadIdx.x;
    const int id0 = input_ids[b];
    for (int k = tid; k < Ee; k += 256)
        xh[(size_t)b * K_C + k] = (bf16_t)embedding[(size_t)id0 * Ee + k];
    for (int k = tid; k < Hh; k += 256) {
        float cv = context[(size_t)b * Hh + k];
        xh[(size_t)b * K_C + Ee + k]        = (bf16_t)cv;
        hc[(size_t)b * 2 * Hh + Hh + k]     = (bf16_t)cv;
        xh[(size_t)b * K_C + K_IH + k]      = (bf16_t)0.0f;   // h0 = 0
        hc[(size_t)b * 2 * Hh + k]          = (bf16_t)0.0f;
        h[(size_t)b * Hh + k] = 0.0f;
        c[(size_t)b * Hh + k] = 0.0f;
    }
}

// Copy enc[:,t,:] (already bf16) into the "aligned" slice of xh.
__global__ void aligned_kernel(const bf16_t* __restrict__ enc_bf,
                               bf16_t* __restrict__ xh, int t)
{
    const int b = blockIdx.x, tid = threadIdx.x;
    const bf16_t* src = enc_bf + ((size_t)b * Tt + t) * Hh;
    bf16_t* dst = xh + (size_t)b * K_C + Ee + Hh;
    for (int k = tid; k < Hh; k += 256) dst[k] = src[k];
}

// ---------------------------------------------------------------------------
// LSTM gates: g = [i f g o] blocks of size H
// ---------------------------------------------------------------------------
__global__ void lstm_kernel(const float* __restrict__ g,
                            float* __restrict__ h, float* __restrict__ c,
                            bf16_t* __restrict__ xh, bf16_t* __restrict__ hc)
{
    int idx = blockIdx.x * 256 + threadIdx.x;       // B*H = 65536 threads
    if (idx >= Bb * Hh) return;
    int b = idx >> 10, j = idx & (Hh - 1);
    const float* gb = g + (size_t)b * G4H;
    float ig = sigf(gb[j]);
    float fg = sigf(gb[Hh + j]);
    float gg = tanhf(gb[2 * Hh + j]);
    float og = sigf(gb[3 * Hh + j]);
    float c2 = fg * c[idx] + ig * gg;
    float h2 = og * tanhf(c2);
    c[idx] = c2;
    h[idx] = h2;
    xh[(size_t)b * K_C + K_IH + j] = (bf16_t)h2;
    hc[(size_t)b * 2 * Hh + j]     = (bf16_t)h2;
}

// ---------------------------------------------------------------------------
// log_softmax over S=128, argmax (first occurrence), embedding gather.
// block = 128 threads, one block per batch row.
// ---------------------------------------------------------------------------
__global__ void slotpost_kernel(const float* __restrict__ score,
                                const float* __restrict__ embedding,
                                float* __restrict__ out, bf16_t* __restrict__ xh,
                                int t)
{
    __shared__ float vmax[Ss]; __shared__ int vidx[Ss]; __shared__ float vsum[Ss];
    const int b = blockIdx.x, s = threadIdx.x;
    const float sc = score[(size_t)b * Ss + s];
    vmax[s] = sc; vidx[s] = s;
    __syncthreads();
    for (int off = 64; off > 0; off >>= 1) {
        if (s < off) {
            float o = vmax[s + off]; int oi = vidx[s + off];
            if (o > vmax[s] || (o == vmax[s] && oi < vidx[s])) { vmax[s] = o; vidx[s] = oi; }
        }
        __syncthreads();
    }
    const float mx = vmax[0];
    vsum[s] = __expf(sc - mx);
    __syncthreads();
    for (int off = 64; off > 0; off >>= 1) {
        if (s < off) vsum[s] += vsum[s + off];
        __syncthreads();
    }
    const float lse = mx + __logf(vsum[0]);
    out[((size_t)b * Tt + t) * Ss + s] = sc - lse;
    const int nx = vidx[0];
    for (int k = s; k < Ee; k += Ss)
        xh[(size_t)b * K_C + k] = (bf16_t)embedding[(size_t)nx * Ee + k];
}

// ---------------------------------------------------------------------------
// Attention: scores = energies . h, softmax over T, ctx = alpha . enc
// block = 256 threads, one block per batch row.
// ---------------------------------------------------------------------------
__global__ void attn_kernel(const float* __restrict__ energies,
                            const float* __restrict__ h,
                            const float* __restrict__ enc,
                            const unsigned char* __restrict__ mask,
                            bf16_t* __restrict__ xh, bf16_t* __restrict__ hc)
{
    __shared__ float part[256]; __shared__ float alpha[Tt];
    const int b = blockIdx.x, tid = threadIdx.x;
    const int tt = tid >> 2, p = tid & 3;
    const float* hb = h + (size_t)b * Hh + p * 256;
    const float* eb = energies + ((size_t)b * Tt + tt) * Hh + p * 256;
    float s = 0.0f;
    for (int k = 0; k < 256; ++k) s += hb[k] * eb[k];
    part[tid] = s;
    __syncthreads();
    if (p == 0) {
        float sc = part[tid] + part[tid + 1] + part[tid + 2] + part[tid + 3];
        alpha[tt] = mask[(size_t)b * Tt + tt] ? NEGV : sc;
    }
    __syncthreads();
    float mx = -3.4e38f;
    for (int t2 = 0; t2 < Tt; ++t2) mx = fmaxf(mx, alpha[t2]);
    float den = 0.0f;
    for (int t2 = 0; t2 < Tt; ++t2) den += __expf(alpha[t2] - mx);
    __syncthreads();
    if (tid < Tt) alpha[tid] = __expf(alpha[tid] - mx) / den;
    __syncthreads();
    const float* encb = enc + (size_t)b * Tt * Hh;
    for (int k = tid; k < Hh; k += 256) {
        float a = 0.0f;
        for (int t2 = 0; t2 < Tt; ++t2) a += alpha[t2] * encb[(size_t)t2 * Hh + k];
        xh[(size_t)b * K_C + Ee + k]    = (bf16_t)a;
        hc[(size_t)b * 2 * Hh + Hh + k] = (bf16_t)a;
    }
}

// ---------------------------------------------------------------------------
// Intent head (h0 == 0 so only the ctx half of intent_W matters, but keep the
// masked-softmax general).
// ---------------------------------------------------------------------------
__global__ void intent_kernel(const float* __restrict__ enc,
                              const unsigned char* __restrict__ mask,
                              const float* __restrict__ iW,
                              const float* __restrict__ ib,
                              float* __restrict__ out_tail)
{
    __shared__ float alpha[Tt]; __shared__ float ctx[Hh]; __shared__ float red[256];
    const int b = blockIdx.x, tid = threadIdx.x;
    if (tid < Tt) alpha[tid] = mask[(size_t)b * Tt + tid] ? NEGV : 0.0f;
    __syncthreads();
    float mx = -3.4e38f;
    for (int t2 = 0; t2 < Tt; ++t2) mx = fmaxf(mx, alpha[t2]);
    float den = 0.0f;
    for (int t2 = 0; t2 < Tt; ++t2) den += __expf(alpha[t2] - mx);
    __syncthreads();
    if (tid < Tt) alpha[tid] = __expf(alpha[tid] - mx) / den;
    __syncthreads();
    for (int k = tid; k < Hh; k += 256) {
        float a = 0.0f;
        for (int t2 = 0; t2 < Tt; ++t2)
            a += alpha[t2] * enc[((size_t)b * Tt + t2) * Hh + k];
        ctx[k] = a;
    }
    __syncthreads();
    const int i = tid >> 3, p = tid & 7;
    float s = 0.0f;
    for (int k = p * 128; k < p * 128 + 128; ++k)
        s += ctx[k] * iW[(size_t)i * (2 * Hh) + Hh + k];
    red[tid] = s;
    __syncthreads();
    if (p == 0) {
        float tot = ib[i];
        for (int q = 0; q < 8; ++q) tot += red[(i << 3) + q];
        out_tail[(size_t)b * Ii + i] = tot;
    }
}

// ---------------------------------------------------------------------------
// Host launcher
// ---------------------------------------------------------------------------
extern "C" void kernel_launch(void* const* d_in, const int* in_sizes, int n_in,
                              void* d_out, int out_size, void* d_ws, size_t ws_size,
                              hipStream_t stream)
{
    const int*   input_ids = (const int*)  d_in[0];
    const float* context   = (const float*)d_in[1];
    const float* enc       = (const float*)d_in[2];
    const unsigned char* mask = (const unsigned char*)d_in[3];
    const float* embedding = (const float*)d_in[4];
    const float* W_ih      = (const float*)d_in[5];
    const float* W_hh      = (const float*)d_in[6];
    const float* b_ih      = (const float*)d_in[7];
    const float* b_hh      = (const float*)d_in[8];
    const float* attn_W    = (const float*)d_in[9];
    const float* attn_b    = (const float*)d_in[10];
    const float* slot_W    = (const float*)d_in[11];
    const float* slot_b    = (const float*)d_in[12];
    const float* intent_W  = (const float*)d_in[13];
    const float* intent_b  = (const float*)d_in[14];
    float* out = (float*)d_out;

    // Workspace carve-up (all 256B aligned).
    char* p = (char*)d_ws;
    auto alloc = [&](size_t bytes) -> char* {
        char* r = p; p += (bytes + 255) & ~(size_t)255; return r;
    };
    bf16_t* Wc       = (bf16_t*)alloc((size_t)G4H * K_C * 2);        // 29.4 MB
    bf16_t* attnW_bf = (bf16_t*)alloc((size_t)Hh * Hh * 2);          //  2.1 MB
    bf16_t* slotW_bf = (bf16_t*)alloc((size_t)Ss * 2 * Hh * 2);      //  0.5 MB
    bf16_t* enc_bf   = (bf16_t*)alloc((size_t)Bb * Tt * Hh * 2);     //  8.4 MB
    float*  energies = (float*) alloc((size_t)Bb * Tt * Hh * 4);     // 16.8 MB
    bf16_t* xh       = (bf16_t*)alloc((size_t)Bb * K_C * 2);
    bf16_t* hc       = (bf16_t*)alloc((size_t)Bb * 2 * Hh * 2);
    float*  g        = (float*) alloc((size_t)Bb * G4H * 4);
    float*  h        = (float*) alloc((size_t)Bb * Hh * 4);
    float*  c        = (float*) alloc((size_t)Bb * Hh * 4);
    float*  score    = (float*) alloc((size_t)Bb * Ss * 4);
    float*  bias_c   = (float*) alloc((size_t)G4H * 4);

    // ---- one-time prep -------------------------------------------------
    build_wc_kernel<<<2048, 256, 0, stream>>>(W_ih, W_hh, Wc);
    bias_kernel<<<16, 256, 0, stream>>>(b_ih, b_hh, bias_c, G4H);
    cvt_kernel<<<1024, 256, 0, stream>>>(attn_W, attnW_bf, (size_t)Hh * Hh);
    cvt_kernel<<<256,  256, 0, stream>>>(slot_W, slotW_bf, (size_t)Ss * 2 * Hh);
    cvt_kernel<<<2048, 256, 0, stream>>>(enc, enc_bf, (size_t)Bb * Tt * Hh);
    init_kernel<<<Bb, 256, 0, stream>>>(input_ids, context, embedding, xh, hc, h, c);

    // energies[B*T][H] = enc @ attn_W^T + attn_b   (M=4096, N=1024, K=1024)
    // 2x2 register blocking: grid (N/(16*2*8)=4, M/32=128)
    gemm_bf16_wmma<2, 2><<<dim3(Hh / 256, (Bb * Tt) / 32), 256, 0, stream>>>(
        enc_bf, Hh, attnW_bf, Hh, attn_b, energies, Hh, Hh);

    intent_kernel<<<Bb, 256, 0, stream>>>(enc, mask, intent_W, intent_b,
                                          out + (size_t)Bb * Tt * Ss);

    // ---- 64 recurrent steps --------------------------------------------
    for (int t = 0; t < Tt; ++t) {
        aligned_kernel<<<Bb, 256, 0, stream>>>(enc_bf, xh, t);
        // g = xh @ [W_ih|W_hh]^T + (b_ih+b_hh)   (M=64, N=4096, K=3584)
        // 2x2 blocking: grid (4096/256=16, 64/32=2) = 32 blocks, 256 waves
        gemm_bf16_wmma<2, 2><<<dim3(G4H / 256, Bb / 32), 256, 0, stream>>>(
            xh, K_C, Wc, K_C, bias_c, g, G4H, K_C);
        lstm_kernel<<<(Bb * Hh) / 256, 256, 0, stream>>>(g, h, c, xh, hc);
        // score = [h|ctx] @ slot_W^T + slot_b   (M=64, N=128, K=2048)
        // tiny GEMM keeps 1x1 tiles so the grid (1,4) stays populated
        gemm_bf16_wmma<1, 1><<<dim3(Ss / 128, Bb / 16), 256, 0, stream>>>(
            hc, 2 * Hh, slotW_bf, 2 * Hh, slot_b, score, Ss, 2 * Hh);
        slotpost_kernel<<<Bb, Ss, 0, stream>>>(score, embedding, out, xh, t);
        attn_kernel<<<Bb, 256, 0, stream>>>(energies, h, enc, mask, xh, hc);
    }
}
